// SCINet_57166014710087
// MI455X (gfx1250) — matrix operand order repaired and checked by reading, not compiled, not run on hardware
//
#include <hip/hip_runtime.h>
#include <hip/hip_bf16.h>
#include <cmath>

// ---------------------------------------------------------------------------
// SCINet forward for MI455X (gfx1250): conv_seq as bf16 WMMA GEMMs.
// ---------------------------------------------------------------------------

typedef __attribute__((ext_vector_type(16))) __bf16 v16bf;
typedef __attribute__((ext_vector_type(8)))  float  v8f;

union ABFrag { v16bf v; uint4 q[2]; };

__device__ __forceinline__ unsigned short f2bf(float f) {
    unsigned int u = __float_as_uint(f);
    u += 0x7FFFu + ((u >> 16) & 1u);      // round-to-nearest-even
    return (unsigned short)(u >> 16);
}

#define TM    30            // output timesteps per workgroup tile
#define XROWS 36            // x rows staged: [t0-6, t0+30)
#define XSTR  136           // 128 + 8 bf16 pad -> 4-bank row skew
#define HROWS 32            // h rows computed: [t0-2, t0+30), 2 M-blocks
#define HSTR  520           // 512 + 8 bf16 pad -> 4-bank row skew
#define CIN   128
#define HID   512

// --- weight pre-pack: f32 -> bf16 in per-lane WMMA B-fragment order --------
// B (32x16 K x N) per lane: column n = lane&15, kk = (e>>3)*16 + (lane>>4)*8 + (e&7), e=0..15
__global__ void SCINet_pack_w1(const float* __restrict__ w,
                               unsigned short* __restrict__ p, int total) {
    int i = blockIdx.x * 256 + threadIdx.x;
    if (i >= total) return;
    int e    = i & 15;
    int lane = (i >> 4) & 31;
    int nb   = (i >> 9) & 31;          // N-block (HID/16 = 32)
    int t    = i >> 14;                // ((is*5 + k)*4 + cb)
    int cb   = t & 3;  t >>= 2;        // K-block of 32 channels (CIN/32 = 4)
    int k    = t % 5;
    int is   = t / 5;                  // (interactor*2 + split)
    int kk   = ((e >> 3) << 4) + ((lane >> 4) << 3) + (e & 7); // 0..31
    int c    = cb * 32 + kk;
    int n    = nb * 16 + (lane & 15);
    size_t src = (((size_t)is * 5 + k) * CIN + c) * (size_t)HID + n;
    p[i] = f2bf(w[src]);
}

__global__ void SCINet_pack_w2(const float* __restrict__ w,
                               unsigned short* __restrict__ p, int total) {
    int i = blockIdx.x * 256 + threadIdx.x;
    if (i >= total) return;
    int e    = i & 15;
    int lane = (i >> 4) & 31;
    int nb   = (i >> 9) & 7;           // N-block (CIN/16 = 8)
    int cb   = (i >> 12) & 15;         // K-block (HID/32 = 16)
    int t    = i >> 16;                // is*3 + k
    int k    = t % 3;
    int is   = t / 3;
    int kk   = ((e >> 3) << 4) + ((lane >> 4) << 3) + (e & 7);
    int c    = cb * 32 + kk;
    int n    = nb * 16 + (lane & 15);
    size_t src = (((size_t)is * 3 + k) * HID + c) * (size_t)CIN + n;
    p[i] = f2bf(w[src]);
}

// --- fused conv_seq: tanh(conv2(leaky(conv1(x)))) + epilogue ---------------
// mode 0: out = other * exp(y)   (mul/gating phase)
// mode 1: out = other + y        (add phase, even split)
// mode 2: out = other - y        (add phase, odd split)
__global__ __launch_bounds__(256) void SCINet_conv_seq(
    const float* __restrict__ x, const float* __restrict__ other,
    float* __restrict__ out,
    const unsigned short* __restrict__ w1p, const float* __restrict__ b1,
    const unsigned short* __restrict__ w2p, const float* __restrict__ b2,
    int L, int mode)
{
    __shared__ __align__(16) unsigned short xs[XROWS * XSTR];
    __shared__ __align__(16) unsigned short hs[(HROWS + 2) * HSTR]; // +2 halo slack

    const int tid  = threadIdx.x;
    const int lane = tid & 31;
    const int wv   = tid >> 5;
    const int r    = lane & 15;   // A row / D column within a 16x16 tile
    const int half = lane >> 4;   // K-half selector
    const int bidx = blockIdx.y;
    const int t0   = blockIdx.x * TM;
    const float* xb = x + (size_t)bidx * L * CIN;

    // Stage x tile (f32 -> bf16) with causal zero-padding.
    for (int i = tid; i < XROWS * CIN; i += 256) {
        int rr = i >> 7, c = i & 127;
        int g = t0 - 6 + rr;
        float v = (g >= 0 && g < L) ? xb[(size_t)g * CIN + c] : 0.0f;
        xs[rr * XSTR + c] = f2bf(v);
    }
    __syncthreads();

    // ---- conv1: h[32 x 512] = leaky(b1 + sum_k x_shift @ W1[k]) -----------
    for (int tile = wv; tile < 2 * (HID / 16); tile += 8) {
        int mb = tile >> 5;        // 0..1
        int nb = tile & 31;        // 0..31
        v8f acc = {};
        for (int k = 0; k < 5; ++k) {
            // h row (rel t0-2) = mb*16 + r  needs x row (rel t0-6) = mb*16 + r + k
            const unsigned short* xr = &xs[(mb * 16 + r + k) * XSTR + half * 8];
            for (int cb = 0; cb < 4; ++cb) {
                ABFrag a, w;
                const uint4* ap = (const uint4*)(xr + cb * 32);
                a.q[0] = ap[0];            // K = half*8 + 0..7
                a.q[1] = ap[2];            // K = half*8 + 16..23
                const uint4* wp =
                    (const uint4*)&w1p[(size_t)((((k * 4 + cb) * 32 + nb) * 32 + lane)) << 4];
                w.q[0] = wp[0];
                w.q[1] = wp[1];
                acc = __builtin_amdgcn_wmma_f32_16x16x32_bf16(
                        false, a.v, false, w.v, (short)0, acc, false, false);
            }
        }
        float bias = b1[nb * 16 + r];
        for (int vr = 0; vr < 8; ++vr) {
            float h = acc[vr] + bias;
            h = h > 0.0f ? h : 0.01f * h;                 // LeakyReLU(0.01)
            hs[(mb * 16 + vr + half * 8) * HSTR + nb * 16 + r] = f2bf(h);
        }
    }
    __syncthreads();

    // ---- conv2 + epilogue: y[32 x 128] = tanh(b2 + sum_k h_shift @ W2[k]) -
    for (int tile = wv; tile < 2 * (CIN / 16); tile += 8) {
        int mb = tile >> 3;        // 0..1
        int nb = tile & 7;         // 0..7
        v8f acc = {};
        for (int k = 0; k < 3; ++k) {
            // y row t0+mb*16+r needs h row (rel t0-2) = mb*16 + r + k
            const unsigned short* hr = &hs[(mb * 16 + r + k) * HSTR + half * 8];
            for (int cb = 0; cb < 16; ++cb) {
                ABFrag a, w;
                const uint4* ap = (const uint4*)(hr + cb * 32);
                a.q[0] = ap[0];
                a.q[1] = ap[2];
                const uint4* wp =
                    (const uint4*)&w2p[(size_t)((((k * 16 + cb) * 8 + nb) * 32 + lane)) << 4];
                w.q[0] = wp[0];
                w.q[1] = wp[1];
                acc = __builtin_amdgcn_wmma_f32_16x16x32_bf16(
                        false, a.v, false, w.v, (short)0, acc, false, false);
            }
        }
        float bias = b2[nb * 16 + r];
        for (int vr = 0; vr < 8; ++vr) {
            int t = t0 + mb * 16 + vr + half * 8;
            if (t < L && (mb * 16 + vr + half * 8) < TM + 2) {
                if (mb * 16 + vr + half * 8 >= TM) continue;   // mask tile tail
                float y = tanhf(acc[vr] + bias);
                size_t oi = ((size_t)bidx * L + t) * CIN + nb * 16 + r;
                float o = other[oi];
                float res = (mode == 0) ? o * __expf(y)
                          : (mode == 1) ? o + y : o - y;
                out[oi] = res;
            }
        }
    }
}

// --- even/odd temporal split ----------------------------------------------
__global__ void SCINet_split(const float* __restrict__ x,
                             float* __restrict__ s0, float* __restrict__ s1,
                             int L, int total) {
    int i = blockIdx.x * 256 + threadIdx.x;
    if (i >= total) return;
    int c  = i & 127;
    int Lh = L >> 1;
    int t  = (i >> 7) % Lh;
    int b  = (i >> 7) / Lh;
    s0[i] = x[((size_t)b * L + 2 * t    ) * CIN + c];
    s1[i] = x[((size_t)b * L + 2 * t + 1) * CIN + c];
}

// --- final rev_split gather: out[b, 512a2+256a1+128a0+w] = leaf[4a0+2a1+a2]
__global__ void SCINet_gather(const float* __restrict__ leaves,
                              float* __restrict__ out, int total) {
    int i = blockIdx.x * 256 + threadIdx.x;
    if (i >= total) return;
    int c   = i & 127;
    int tau = (i >> 7) & 1023;
    int b   = i >> 17;
    int w   = tau & 127;
    int a0  = (tau >> 7) & 1;
    int a1  = (tau >> 8) & 1;
    int a2  = tau >> 9;
    int m   = 4 * a0 + 2 * a1 + a2;
    out[i] = leaves[(((size_t)m * 32 + b) * 128 + w) * CIN + c];
}

// ---------------------------------------------------------------------------
extern "C" void kernel_launch(void* const* d_in, const int* in_sizes, int n_in,
                              void* d_out, int out_size, void* d_ws, size_t ws_size,
                              hipStream_t stream) {
    const float* x      = (const float*)d_in[0];
    const float* mul_w1 = (const float*)d_in[1];
    const float* mul_b1 = (const float*)d_in[2];
    const float* mul_w2 = (const float*)d_in[3];
    const float* mul_b2 = (const float*)d_in[4];
    const float* add_w1 = (const float*)d_in[5];
    const float* add_b1 = (const float*)d_in[6];
    const float* add_w2 = (const float*)d_in[7];
    const float* add_b2 = (const float*)d_in[8];

    const size_t W1SL = 5ull * CIN * HID;       // 327680 bf16 per (idx,split)
    const size_t W2SL = 3ull * HID * CIN;       // 196608 bf16 per (idx,split)
    const size_t W1E  = 14ull * W1SL;
    const size_t W2E  = 14ull * W2SL;
    const size_t SZ1  = 32ull * 512 * 128;
    const size_t SZ2  = 32ull * 256 * 128;
    const size_t SZ3  = 32ull * 128 * 128;

    char* ws = (char*)d_ws;
    size_t off = 0;
    auto carve = [&](size_t bytes) -> void* {
        void* p = ws + off;
        off = (off + bytes + 255) & ~(size_t)255;
        return p;
    };
    unsigned short* pmw1 = (unsigned short*)carve(W1E * 2);
    unsigned short* paw1 = (unsigned short*)carve(W1E * 2);
    unsigned short* pmw2 = (unsigned short*)carve(W2E * 2);
    unsigned short* paw2 = (unsigned short*)carve(W2E * 2);
    float* n1 = (float*)carve(2 * SZ1 * 4);
    float* n2 = (float*)carve(4 * SZ2 * 4);
    float* n3 = (float*)carve(8 * SZ3 * 4);
    float* S0 = (float*)carve(SZ1 * 4);
    float* S1 = (float*)carve(SZ1 * 4);
    float* M0 = (float*)carve(SZ1 * 4);
    float* M1 = (float*)carve(SZ1 * 4);

    // Pack all weights to bf16 WMMA fragment layout (L2-resident afterwards).
    SCINet_pack_w1<<<(int)((W1E + 255) / 256), 256, 0, stream>>>(mul_w1, pmw1, (int)W1E);
    SCINet_pack_w1<<<(int)((W1E + 255) / 256), 256, 0, stream>>>(add_w1, paw1, (int)W1E);
    SCINet_pack_w2<<<(int)((W2E + 255) / 256), 256, 0, stream>>>(mul_w2, pmw2, (int)W2E);
    SCINet_pack_w2<<<(int)((W2E + 255) / 256), 256, 0, stream>>>(add_w2, paw2, (int)W2E);

    auto conv = [&](const float* xin, const float* oth, float* o,
                    const unsigned short* w1, const float* bb1,
                    const unsigned short* w2, const float* bb2, int L, int mode) {
        dim3 g((L + TM - 1) / TM, 32);
        SCINet_conv_seq<<<g, 256, 0, stream>>>(xin, oth, o, w1, bb1, w2, bb2, L, mode);
    };

    auto interact = [&](const float* X, int L, float* c0, float* c1, int idx) {
        int Lh  = L / 2;
        int tot = 32 * Lh * CIN;
        SCINet_split<<<(tot + 255) / 256, 256, 0, stream>>>(X, S0, S1, L, tot);
        int i0 = idx * 2, i1 = idx * 2 + 1;
        // mul phase: M_i = S_i * exp(conv_seq(S_{1-i}, mul[idx,i]))
        conv(S1, S0, M0, pmw1 + (size_t)i0 * W1SL, mul_b1 + i0 * HID,
                         pmw2 + (size_t)i0 * W2SL, mul_b2 + i0 * CIN, Lh, 0);
        conv(S0, S1, M1, pmw1 + (size_t)i1 * W1SL, mul_b1 + i1 * HID,
                         pmw2 + (size_t)i1 * W2SL, mul_b2 + i1 * CIN, Lh, 0);
        // add phase: out0 = M0 + conv(M1), out1 = M1 - conv(M0)
        conv(M1, M0, c0, paw1 + (size_t)i0 * W1SL, add_b1 + i0 * HID,
                         paw2 + (size_t)i0 * W2SL, add_b2 + i0 * CIN, Lh, 1);
        conv(M0, M1, c1, paw1 + (size_t)i1 * W1SL, add_b1 + i1 * HID,
                         paw2 + (size_t)i1 * W2SL, add_b2 + i1 * CIN, Lh, 2);
    };

    // level 0
    interact(x, 1024, n1, n1 + SZ1, 0);
    // level 1
    interact(n1,       512, n2,           n2 + SZ2,     1);
    interact(n1 + SZ1, 512, n2 + 2 * SZ2, n2 + 3 * SZ2, 2);
    // level 2
    for (int j = 0; j < 4; ++j)
        interact(n2 + (size_t)j * SZ2, 256,
                 n3 + (size_t)(2 * j) * SZ3, n3 + (size_t)(2 * j + 1) * SZ3, 3 + j);

    int totO = 32 * 1024 * CIN;
    SCINet_gather<<<(totO + 255) / 256, 256, 0, stream>>>(n3, (float*)d_out, totO);
}
